// TalkingHeadAttention_43928925503607
// MI455X (gfx1250) — matrix operand order repaired and checked by reading, not compiled, not run on hardware
//
#include <hip/hip_runtime.h>
#include <hip/hip_bf16.h>
#include <stdint.h>

#define N_TOK 2048
#define DIM   1024
#define NH    16
#define DKH   64
#define TOPK_ 128
#define LN_EPS 1e-5f
#define SPITCH 2052   // LDS row pitch (floats): 2052%64=4 -> 16 head reads hit 16 banks

typedef __bf16 bf16;
typedef __attribute__((ext_vector_type(16))) __bf16 v16bf;
typedef __attribute__((ext_vector_type(8)))  float  v8f;
typedef __attribute__((ext_vector_type(4)))  unsigned v4u;
typedef __attribute__((ext_vector_type(8)))  int    v8i;
typedef __attribute__((ext_vector_type(4)))  int    v4i;

union FragBF { v16bf v; uint4 u[2]; };

#if __has_builtin(__builtin_amdgcn_tensor_load_to_lds)
  #define HAVE_TDM 1
  #if __has_include(<hip/amd_detail/amd_gfx1250_TDM.h>)
    #define TDM_6ARG 1   // therock/clang-23 arity
  #endif
#endif

// ---------------- wave32 reductions ----------------
__device__ __forceinline__ float wave_max_f(float v){
#pragma unroll
  for (int o = 16; o > 0; o >>= 1) v = fmaxf(v, __shfl_xor(v, o, 32));
  return v;
}
__device__ __forceinline__ float wave_sum_f(float v){
#pragma unroll
  for (int o = 16; o > 0; o >>= 1) v += __shfl_xor(v, o, 32);
  return v;
}
__device__ __forceinline__ int wave_sum_i(int v){
#pragma unroll
  for (int o = 16; o > 0; o >>= 1) v += __shfl_xor(v, o, 32);
  return v;
}

#if HAVE_TDM
// ---- TDM: DMA one 128x32 bf16 tile (row stride = ld elements) into LDS ----
__device__ __forceinline__ void tdm_load_tile_128x32(const bf16* gtile, unsigned lds_byte_off,
                                                     long ld_elems){
  const unsigned long long ga = (unsigned long long)(size_t)gtile;
  v4u g0;
  g0.x = 1u;                                             // count=1 (valid user D#)
  g0.y = lds_byte_off;                                   // lds_addr [63:32]
  g0.z = (unsigned)(ga & 0xFFFFFFFFull);                 // global_addr[31:0]
  g0.w = (unsigned)((ga >> 32) & 0x01FFFFFFull) | (2u << 30); // addr[56:32] | type=2
  const unsigned long long st = (unsigned long long)ld_elems;
  v8i g1;
  g1[0] = (int)(1u << 16);                               // wg_mask=0, data_size=1 (2B)
  g1[1] = (int)(32u << 16);                              // tensor_dim0[15:0]=32 in [63:48]
  g1[2] = (int)(0u | (128u << 16));                      // dim0 hi=0 | tensor_dim1 lo=128
  g1[3] = (int)(0u | (32u << 16));                       // dim1 hi=0 | tile_dim0=32
  g1[4] = (int)(128u);                                   // tile_dim1=128, tile_dim2=0
  g1[5] = (int)(st & 0xFFFFFFFFull);                     // tensor_dim0_stride[31:0]
  g1[6] = (int)((st >> 32) & 0xFFFFull);                 // stride[47:32] | dim1_stride lo=0
  g1[7] = 0;
  const v4i z4 = {0, 0, 0, 0};
#if defined(TDM_6ARG)
  const v8i z8 = {0, 0, 0, 0, 0, 0, 0, 0};
  __builtin_amdgcn_tensor_load_to_lds(g0, g1, z4, z4, z8, 0);
#else
  __builtin_amdgcn_tensor_load_to_lds(g0, g1, z4, z4, 0);
#endif
}
#endif

__device__ __forceinline__ unsigned lds_offset_of(const void* p){
  // generic (flat) LDS addresses carry the wave-relative LDS byte offset in the low 32 bits
  return (unsigned)(size_t)p;
}

// ---------------- fp32 -> bf16 convert (weights) ----------------
__global__ void cvt_f32_bf16(const float* __restrict__ s, bf16* __restrict__ d, long n){
  long i = (long)blockIdx.x * blockDim.x + threadIdx.x;
  long stride = (long)gridDim.x * blockDim.x;
  for (; i < n; i += stride) d[i] = (bf16)s[i];
}

// ---------------- LayerNorm -> bf16 ----------------
__global__ __launch_bounds__(256)
void ln_bf16(const float* __restrict__ x, const float* __restrict__ gamma,
             const float* __restrict__ beta, bf16* __restrict__ xn){
  __shared__ float red[256];
  const int row = blockIdx.x, tid = threadIdx.x;
  const float4 v = ((const float4*)(x + (long)row * DIM))[tid];
  red[tid] = v.x + v.y + v.z + v.w;
  __syncthreads();
  for (int o = 128; o > 0; o >>= 1){ if (tid < o) red[tid] += red[tid + o]; __syncthreads(); }
  const float mu = red[0] * (1.0f / DIM);
  __syncthreads();
  const float dx = v.x - mu, dy = v.y - mu, dz = v.z - mu, dw = v.w - mu;
  red[tid] = dx*dx + dy*dy + dz*dz + dw*dw;
  __syncthreads();
  for (int o = 128; o > 0; o >>= 1){ if (tid < o) red[tid] += red[tid + o]; __syncthreads(); }
  const float rs = rsqrtf(red[0] * (1.0f / DIM) + LN_EPS);
  const float4 gm = ((const float4*)gamma)[tid];
  const float4 bt = ((const float4*)beta)[tid];
  bf16* o = xn + (long)row * DIM + tid * 4;
  o[0] = (bf16)(dx * rs * gm.x + bt.x);
  o[1] = (bf16)(dy * rs * gm.y + bt.y);
  o[2] = (bf16)(dz * rs * gm.z + bt.z);
  o[3] = (bf16)(dw * rs * gm.w + bt.w);
}

// ---------------- q' = scale * Wpre[g,h] * q  (folds pre-mix into Q) ----------------
__global__ void make_qprime(const bf16* __restrict__ q, const float* __restrict__ Wpre,
                            bf16* __restrict__ qp){
  const float scale = 0.125f; // 1/sqrt(DK)
  const long total = (long)NH * N_TOK * DIM;
  long i = (long)blockIdx.x * blockDim.x + threadIdx.x;
  const long stride = (long)gridDim.x * blockDim.x;
  for (; i < total; i += stride){
    const int c = (int)(i & (DIM - 1));
    const long rem = i >> 10;
    const int n = (int)(rem & (N_TOK - 1));
    const int g = (int)(rem >> 11);
    const int h = c >> 6;
    qp[i] = (bf16)((float)q[(long)n * DIM + c] * scale * Wpre[g * NH + h]);
  }
}

// ---------------- tiled bf16 transpose (V -> V^T) ----------------
__global__ __launch_bounds__(256)
void transpose_bf16(const bf16* __restrict__ in, bf16* __restrict__ out, int rows, int cols){
  __shared__ bf16 tile[32][33];
  const int tx = threadIdx.x, ty = threadIdx.y;
  const int c0 = blockIdx.x * 32, r0 = blockIdx.y * 32;
#pragma unroll
  for (int j = 0; j < 32; j += 8) tile[ty + j][tx] = in[(long)(r0 + ty + j) * cols + c0 + tx];
  __syncthreads();
#pragma unroll
  for (int j = 0; j < 32; j += 8) out[(long)(c0 + ty + j) * rows + r0 + tx] = tile[tx][ty + j];
}

// =========================================================================
// Macro-tiled NT GEMM: 128x128 C tile per workgroup, TDM-fed double-buffered
// LDS staging, 8 waves each computing a 32x64 register block (2x4 WMMA tiles).
// C[m,n] = sum_k A[m,k]*B[n,k] (+bias)(+residual). M%128==N%128==K%32==0.
// =========================================================================
__global__ __launch_bounds__(256)
void gemm_nt_tdm(const bf16* __restrict__ A, long lda, long sA,
                 const bf16* __restrict__ B, long ldb, long sB,
                 void* __restrict__ Cv, long ldc, long sC,
                 int M, int Ncols, int K,
                 const float* __restrict__ bias, int bias_mode,
                 const float* __restrict__ resid, long ldr,
                 int out_bf16){
  __shared__ bf16 lsA[2][128 * 32];
  __shared__ bf16 lsB[2][128 * 32];
  const int tid  = threadIdx.x;
  const int lane = tid & 31, wave = tid >> 5;
  const int l = lane & 15, hi = lane >> 4;
  const int wm = wave & 3;        // 4 waves along M (4*32 = 128)
  const int wn = wave >> 2;       // 2 waves along N (2*64 = 128)
  const int batch = blockIdx.z;
  const long m0 = (long)blockIdx.y * 128;
  const long n0 = (long)blockIdx.x * 128;
  const bf16* Ab = A + (long)batch * sA + m0 * lda;
  const bf16* Bb = B + (long)batch * sB + n0 * ldb;
  const int nk = K >> 5;

  v8f acc[2][4];
#pragma unroll
  for (int i = 0; i < 2; ++i)
#pragma unroll
    for (int j = 0; j < 4; ++j)
#pragma unroll
      for (int r = 0; r < 8; ++r) acc[i][j][r] = 0.f;

#if HAVE_TDM
  if (wave == 0){
    tdm_load_tile_128x32(Ab, lds_offset_of(&lsA[0][0]), lda);
    tdm_load_tile_128x32(Bb, lds_offset_of(&lsB[0][0]), ldb);
  }
  for (int kt = 0; kt < nk; ++kt){
    const int cur = kt & 1;
    if (wave == 0){
      if (kt + 1 < nk){
        // prefetch next k-slice into the other buffer, then wait only for current
        tdm_load_tile_128x32(Ab + (long)(kt + 1) * 32, lds_offset_of(&lsA[cur ^ 1][0]), lda);
        tdm_load_tile_128x32(Bb + (long)(kt + 1) * 32, lds_offset_of(&lsB[cur ^ 1][0]), ldb);
        __builtin_amdgcn_s_wait_tensorcnt((short)2);
      } else {
        __builtin_amdgcn_s_wait_tensorcnt((short)0);
      }
    }
    __syncthreads();                         // current tile visible to all waves
    const bf16* As = &lsA[cur][0];
    const bf16* Bs = &lsB[cur][0];
#else
  for (int kt = 0; kt < nk; ++kt){
    __syncthreads();                         // previous compute done before overwrite
    {  // cooperative fallback copy: 256 threads, 16 elems (32B) each per matrix
      const int r = tid >> 1, h = tid & 1;
      const uint4* gA = (const uint4*)(Ab + (long)r * lda + (long)kt * 32 + h * 16);
      uint4* dA = (uint4*)(&lsA[0][r * 32 + h * 16]);
      dA[0] = gA[0]; dA[1] = gA[1];
      const uint4* gB = (const uint4*)(Bb + (long)r * ldb + (long)kt * 32 + h * 16);
      uint4* dB = (uint4*)(&lsB[0][r * 32 + h * 16]);
      dB[0] = gB[0]; dB[1] = gB[1];
    }
    __syncthreads();
    const bf16* As = &lsA[0][0];
    const bf16* Bs = &lsB[0][0];
#endif
    FragBF fa[2], fb[4];
#pragma unroll
    for (int i = 0; i < 2; ++i){
      const bf16* ap = As + (wm * 32 + i * 16 + l) * 32 + hi * 8;
      fa[i].u[0] = *(const uint4*)(ap);
      fa[i].u[1] = *(const uint4*)(ap + 16);
    }
#pragma unroll
    for (int j = 0; j < 4; ++j){
      const bf16* bp = Bs + (wn * 64 + j * 16 + l) * 32 + hi * 8;
      fb[j].u[0] = *(const uint4*)(bp);
      fb[j].u[1] = *(const uint4*)(bp + 16);
    }
#pragma unroll
    for (int i = 0; i < 2; ++i)
#pragma unroll
      for (int j = 0; j < 4; ++j)
        acc[i][j] = __builtin_amdgcn_wmma_f32_16x16x32_bf16(false, fa[i].v, false, fb[j].v,
                                                            (short)0, acc[i][j], false, false);
    __syncthreads();                         // done reading before next overwrite
  }

#pragma unroll
  for (int i = 0; i < 2; ++i){
#pragma unroll
    for (int j = 0; j < 4; ++j){
#pragma unroll
      for (int r = 0; r < 8; ++r){
        const long row = m0 + wm * 32 + i * 16 + hi * 8 + r;
        const long col = n0 + wn * 64 + j * 16 + l;
        float v = acc[i][j][r];
        if (bias_mode == 1) v += bias[col];
        else if (bias_mode == 2) v += bias[batch];
        if (resid) v += resid[row * ldr + col];
        const long off = (long)batch * sC + row * ldc + col;
        if (out_bf16) ((bf16*)Cv)[off] = (bf16)v;
        else          ((float*)Cv)[off] = v;
      }
    }
  }
}

// ---------------- simple direct-global NT GEMM (used for AV: N=64) ----------------
__global__ __launch_bounds__(256)
void gemm_nt_wmma(const bf16* __restrict__ A, long lda, long sA,
                  const bf16* __restrict__ B, long ldb, long sB,
                  void* __restrict__ Cv, long ldc, long sC,
                  int M, int Ncols, int K,
                  const float* __restrict__ bias, int bias_mode,
                  const float* __restrict__ resid, long ldr,
                  int out_bf16){
  const int lane = threadIdx.x & 31;
  const int wave = threadIdx.x >> 5;
  const int l  = lane & 15;
  const int hi = lane >> 4;
  const int batch = blockIdx.z;
  const int n0 = (blockIdx.x * 8 + wave) * 16;
  const int m0 = blockIdx.y * 16;
  if (n0 >= Ncols || m0 >= M) return;   // wave-uniform: EXEC stays all-1 for WMMA

  const bf16* a0 = A + (long)batch * sA + (long)(m0 + l) * lda;
  const bf16* b0 = B + (long)batch * sB + (long)(n0 + l) * ldb;

  v8f c = {0.f, 0.f, 0.f, 0.f, 0.f, 0.f, 0.f, 0.f};
#pragma unroll 2
  for (int k0 = 0; k0 < K; k0 += 32){
    FragBF fa, fb;
    const bf16* ap = a0 + k0 + hi * 8;
    fa.u[0] = *(const uint4*)(ap);
    fa.u[1] = *(const uint4*)(ap + 16);
    const bf16* bp = b0 + k0 + hi * 8;
    fb.u[0] = *(const uint4*)(bp);
    fb.u[1] = *(const uint4*)(bp + 16);
    c = __builtin_amdgcn_wmma_f32_16x16x32_bf16(false, fa.v, false, fb.v,
                                                (short)0, c, false, false);
  }
#pragma unroll
  for (int r = 0; r < 8; ++r){
    const int row = m0 + hi * 8 + r;
    const int col = n0 + l;
    float v = c[r];
    if (bias_mode == 1) v += bias[col];
    else if (bias_mode == 2) v += bias[batch];
    if (resid) v += resid[(long)row * ldr + col];
    const long off = (long)batch * sC + (long)row * ldc + col;
    if (out_bf16) ((bf16*)Cv)[off] = (bf16)v;
    else          ((float*)Cv)[off] = v;
  }
}

// ---------------- fused softmax + exact top-k (bisection) + post-mix ----------------
__global__ __launch_bounds__(512)
void softmax_topk_postmix(const float* __restrict__ S,
                          const float* __restrict__ Wpost,
                          const float* __restrict__ bpost,
                          bf16* __restrict__ A2){
  extern __shared__ __align__(16) float lds[];
  float* rowbuf = lds;                 // 16 * SPITCH
  float* wl = lds + 16 * SPITCH;       // 256
  float* bl = wl + 256;                // 16
  const int tid = threadIdx.x;
  if (tid < 256) wl[tid] = Wpost[tid];
  if (tid < 16)  bl[tid] = bpost[tid];

  const int q = blockIdx.x;
  const int g = tid >> 5;
  const int lane = tid & 31;
  const float4* s4 = (const float4*)(S + ((long)g * N_TOK + q) * N_TOK);

  float p[64];
#pragma unroll
  for (int i = 0; i < 16; ++i){
    float4 t = s4[i * 32 + lane];
    p[4*i+0] = t.x; p[4*i+1] = t.y; p[4*i+2] = t.z; p[4*i+3] = t.w;
  }
  float m = -3.4e38f;
#pragma unroll
  for (int i = 0; i < 64; ++i) m = fmaxf(m, p[i]);
  m = wave_max_f(m);
  float zs = 0.f;
#pragma unroll
  for (int i = 0; i < 64; ++i){ p[i] = __expf(p[i] - m); zs += p[i]; }
  const float Z = wave_sum_f(zs);

  // bisection for the 128th-largest of p (max p == 1 by construction)
  float lo = 0.f, hif = 1.0f;
  for (int it = 0; it < 30; ++it){
    const float mid = 0.5f * (lo + hif);
    int cnt = 0;
#pragma unroll
    for (int i = 0; i < 64; ++i) cnt += (p[i] >= mid);
    cnt = wave_sum_i(cnt);
    if (cnt >= TOPK_) lo = mid; else hif = mid;
  }
  const float thr = lo;
  float ms = 0.f;
#pragma unroll
  for (int i = 0; i < 64; ++i){ p[i] = (p[i] >= thr) ? p[i] : 0.f; ms += p[i]; }
  ms = wave_sum_f(ms);
  const float inv = 1.0f / (ms + 1e-9f * Z);   // == attn/(sum+1e-9) scaled through Z

  float4* dst4 = (float4*)(rowbuf + g * SPITCH);
#pragma unroll
  for (int i = 0; i < 16; ++i){
    float4 t4;
    t4.x = p[4*i+0]*inv; t4.y = p[4*i+1]*inv; t4.z = p[4*i+2]*inv; t4.w = p[4*i+3]*inv;
    dst4[i * 32 + lane] = t4;
  }
  __syncthreads();

  // post-softmax talking heads: attn''[g2,q,k] = bpost[g2] + sum_g Wpost[g2,g]*attn'[g,q,k]
  for (int j = 0; j < 4; ++j){
    const int k = tid + 512 * j;
    float a[16];
#pragma unroll
    for (int gg = 0; gg < 16; ++gg) a[gg] = rowbuf[gg * SPITCH + k];
#pragma unroll
    for (int g2 = 0; g2 < 16; ++g2){
      float acc = bl[g2];
#pragma unroll
      for (int gg = 0; gg < 16; ++gg) acc += wl[g2 * 16 + gg] * a[gg];
      A2[((long)g2 * N_TOK + q) * N_TOK + k] = (bf16)acc;
    }
  }
}

// ---------------- host launcher ----------------
extern "C" void kernel_launch(void* const* d_in, const int* in_sizes, int n_in,
                              void* d_out, int out_size, void* d_ws, size_t ws_size,
                              hipStream_t stream){
  const float* x     = (const float*)d_in[0];
  const float* Wq    = (const float*)d_in[1];
  const float* bq    = (const float*)d_in[2];
  const float* Wk    = (const float*)d_in[3];
  const float* bk    = (const float*)d_in[4];
  const float* Wv    = (const float*)d_in[5];
  const float* bv    = (const float*)d_in[6];
  const float* Wpre  = (const float*)d_in[7];
  const float* bpre  = (const float*)d_in[8];   // constant shift along k: cancels in softmax
  const float* Wpost = (const float*)d_in[9];
  const float* bpost = (const float*)d_in[10];
  const float* Wo    = (const float*)d_in[11];
  const float* bo    = (const float*)d_in[12];
  const float* gamma = (const float*)d_in[13];
  const float* beta  = (const float*)d_in[14];

  char* w = (char*)d_ws;
  size_t off = 0;
  auto alloc = [&](size_t bytes) -> char* {
    char* p = w + off;
    off = (off + bytes + 255) & ~(size_t)255;
    return p;
  };
  bf16* XN  = (bf16*)alloc((size_t)N_TOK * DIM * 2);
  bf16* WQB = (bf16*)alloc((size_t)DIM * DIM * 2);
  bf16* WKB = (bf16*)alloc((size_t)DIM * DIM * 2);
  bf16* WVB = (bf16*)alloc((size_t)DIM * DIM * 2);
  bf16* WOB = (bf16*)alloc((size_t)DIM * DIM * 2);
  bf16* Qb  = (bf16*)alloc((size_t)N_TOK * DIM * 2);
  bf16* Kb  = (bf16*)alloc((size_t)N_TOK * DIM * 2);
  bf16* Vb  = (bf16*)alloc((size_t)N_TOK * DIM * 2);
  bf16* VT  = (bf16*)alloc((size_t)DIM * N_TOK * 2);
  bf16* QP  = (bf16*)alloc((size_t)NH * N_TOK * DIM * 2);
  float* Sb = (float*)alloc((size_t)NH * N_TOK * N_TOK * 4);
  bf16* A2  = (bf16*)alloc((size_t)NH * N_TOK * N_TOK * 2);
  bf16* CTX = (bf16*)alloc((size_t)N_TOK * DIM * 2);
  (void)ws_size; (void)in_sizes; (void)n_in; (void)out_size;

  const long WN = (long)DIM * DIM;
  cvt_f32_bf16<<<1024, 256, 0, stream>>>(Wq, WQB, WN);
  cvt_f32_bf16<<<1024, 256, 0, stream>>>(Wk, WKB, WN);
  cvt_f32_bf16<<<1024, 256, 0, stream>>>(Wv, WVB, WN);
  cvt_f32_bf16<<<1024, 256, 0, stream>>>(Wo, WOB, WN);

  ln_bf16<<<N_TOK, 256, 0, stream>>>(x, gamma, beta, XN);

  // Q/K/V projections: xn (2048x1024) @ W^T (1024x1024) -> bf16 (TDM macro-tiled)
  const dim3 gProj(DIM / 128, N_TOK / 128, 1);
  gemm_nt_tdm<<<gProj, 256, 0, stream>>>(XN, DIM, 0, WQB, DIM, 0, Qb, DIM, 0,
                                         N_TOK, DIM, DIM, bq, 1, nullptr, 0, 1);
  gemm_nt_tdm<<<gProj, 256, 0, stream>>>(XN, DIM, 0, WKB, DIM, 0, Kb, DIM, 0,
                                         N_TOK, DIM, DIM, bk, 1, nullptr, 0, 1);
  gemm_nt_tdm<<<gProj, 256, 0, stream>>>(XN, DIM, 0, WVB, DIM, 0, Vb, DIM, 0,
                                         N_TOK, DIM, DIM, bv, 1, nullptr, 0, 1);

  make_qprime<<<8192, 256, 0, stream>>>(Qb, Wpre, QP);
  transpose_bf16<<<dim3(DIM / 32, N_TOK / 32), dim3(32, 8), 0, stream>>>(Vb, VT, N_TOK, DIM);

  // mixed scores: S[g] = q'[g] (2048x1024) @ K^T + bpre[g] -> fp32, batched over 16 heads
  const dim3 gScores(N_TOK / 128, N_TOK / 128, NH);
  gemm_nt_tdm<<<gScores, 256, 0, stream>>>(QP, DIM, (long)N_TOK * DIM,
                                           Kb, DIM, 0,
                                           Sb, N_TOK, (long)N_TOK * N_TOK,
                                           N_TOK, N_TOK, DIM, bpre, 2, nullptr, 0, 0);

  // softmax + top-128 + post-mix (one WG per query; 16 waves = 16 heads)
  const size_t smem = (size_t)(16 * SPITCH + 256 + 16) * 4;
  softmax_topk_postmix<<<N_TOK, 512, smem, stream>>>(Sb, Wpost, bpost, A2);

  // AV: ctx[:, g2*64:(g2+1)*64] = attn''[g2] (2048x2048) @ vT[g2*64:+64, :]^T
  const dim3 gAV(1, N_TOK / 16, NH);
  gemm_nt_wmma<<<gAV, 256, 0, stream>>>(A2, N_TOK, (long)N_TOK * N_TOK,
                                        VT, N_TOK, (long)DKH * N_TOK,
                                        CTX, DIM, DKH,
                                        N_TOK, DKH, N_TOK, nullptr, 0, nullptr, 0, 1);

  // final: out = x + ctx @ Wo^T + bo  (fp32 out, TDM macro-tiled)
  gemm_nt_tdm<<<gProj, 256, 0, stream>>>(CTX, DIM, 0, WOB, DIM, 0, d_out, DIM, 0,
                                         N_TOK, DIM, DIM, bo, 1, x, DIM, 0);
}
